// CuGraphSAGE_9277129360068
// MI455X (gfx1250) — matrix operand order repaired and checked by reading, not compiled, not run on hardware
//
#include <hip/hip_runtime.h>
#include <hip/hip_bf16.h>

// ---------------------------------------------------------------------------
// GraphSAGE (3 layers + linear head) for MI455X / gfx1250.
//  - edge aggregation: wave-per-edge, float4 per lane, global_atomic_add_f32
//  - GEMMs: v_wmma_f32_16x16x32_bf16, operands pre-packed in ISA fragment
//    layout so each lane loads one contiguous 32B chunk per WMMA.
// ---------------------------------------------------------------------------

typedef __attribute__((ext_vector_type(16))) __bf16        v16bf;
typedef __attribute__((ext_vector_type(8)))  float         v8f;
typedef __attribute__((ext_vector_type(16))) unsigned short v16u;

__device__ __forceinline__ unsigned short f2bf(float f) {
  // round-to-nearest-even f32 -> bf16
  unsigned u = __builtin_bit_cast(unsigned, f);
  u += 0x7FFFu + ((u >> 16) & 1u);
  return (unsigned short)(u >> 16);
}

__device__ __forceinline__ unsigned hash32(unsigned x) {
  x ^= x >> 16; x *= 0x7feb352du;
  x ^= x >> 15; x *= 0x846ca68bu;
  x ^= x >> 16;
  return x;
}

// ---------------------------------------------------------------------------
// Degree accumulation: deg[dst] += 1
// ---------------------------------------------------------------------------
__global__ void deg_kernel(const int* __restrict__ dst, float* __restrict__ degf, int E) {
  int e = blockIdx.x * blockDim.x + threadIdx.x;
  if (e < E) {
    __hip_atomic_fetch_add(&degf[dst[e]], 1.0f,
                           __ATOMIC_RELAXED, __HIP_MEMORY_SCOPE_AGENT);
  }
}

// ---------------------------------------------------------------------------
// Scatter-add: one wave per edge, each lane handles 4 features (float4).
// agg[dst] += h[src]   (128 features)
// ---------------------------------------------------------------------------
__global__ void scatter_kernel(const float* __restrict__ h,
                               const int* __restrict__ src,
                               const int* __restrict__ dst,
                               float* __restrict__ agg, int E) {
  unsigned gid = blockIdx.x * blockDim.x + threadIdx.x;
  unsigned e = gid >> 5;      // edge index (uniform per wave)
  unsigned q = gid & 31;      // float4 slot: 32 * 4 = 128 features
  if (e >= (unsigned)E) return;
  int s = src[e];
  int d = dst[e];
  const float4* hp = reinterpret_cast<const float4*>(h + (size_t)s * 128);
  float4 v = hp[q];
  float* ap = agg + (size_t)d * 128 + q * 4;
  __hip_atomic_fetch_add(ap + 0, v.x, __ATOMIC_RELAXED, __HIP_MEMORY_SCOPE_AGENT);
  __hip_atomic_fetch_add(ap + 1, v.y, __ATOMIC_RELAXED, __HIP_MEMORY_SCOPE_AGENT);
  __hip_atomic_fetch_add(ap + 2, v.z, __ATOMIC_RELAXED, __HIP_MEMORY_SCOPE_AGENT);
  __hip_atomic_fetch_add(ap + 3, v.w, __ATOMIC_RELAXED, __HIP_MEMORY_SCOPE_AGENT);
}

// ---------------------------------------------------------------------------
// Pack A-matrix fragments (bf16) in the exact WMMA 16-bit A 16x32 layout:
//   lane l holds row (l%16); element j -> K = (l/16)*8 + (j<8 ? j : j+8)
// mode==0: K in [0,128) -> agg[row][K]/max(deg,1); K in [128,256) -> h[row][K-128]
// mode==1: K in [0,128) -> h[row][K]  (head GEMM)
// Packed offset for thread t (one 32B group): t = (mb*numKb + kb)*32 + lane
// ---------------------------------------------------------------------------
__global__ void pack_a_kernel(const float* __restrict__ agg,
                              const float* __restrict__ degf,
                              const float* __restrict__ h,
                              unsigned short* __restrict__ outp,
                              int N, int numKb, int mode) {
  int Mb = (N + 15) >> 4;
  unsigned t = blockIdx.x * blockDim.x + threadIdx.x;
  unsigned total = (unsigned)Mb * numKb * 32u;
  if (t >= total) return;
  unsigned lane = t & 31u;
  unsigned kb = (t >> 5) % (unsigned)numKb;
  unsigned mb = t / (32u * numKb);
  unsigned m = lane & 15u;
  unsigned half = lane >> 4;
  unsigned row = mb * 16u + m;

  unsigned short o[16];
  if (row < (unsigned)N) {
    const float* hrow = h + (size_t)row * 128;
    const float* arow = (mode == 0) ? (agg + (size_t)row * 128) : nullptr;
    float inv = (mode == 0) ? (1.0f / fmaxf(degf[row], 1.0f)) : 0.0f;
#pragma unroll
    for (int j = 0; j < 16; ++j) {
      unsigned K = kb * 32u + half * 8u + (unsigned)((j < 8) ? j : j + 8);
      float v;
      if (mode == 0) v = (K < 128u) ? arow[K] * inv : hrow[K - 128u];
      else           v = hrow[K];
      o[j] = f2bf(v);
    }
  } else {
#pragma unroll
    for (int j = 0; j < 16; ++j) o[j] = 0;
  }
  unsigned rw[8];
#pragma unroll
  for (int j = 0; j < 8; ++j)
    rw[j] = (unsigned)o[2 * j] | ((unsigned)o[2 * j + 1] << 16);
  uint4* dst4 = reinterpret_cast<uint4*>(outp + (size_t)t * 16);
  dst4[0] = make_uint4(rw[0], rw[1], rw[2], rw[3]);
  dst4[1] = make_uint4(rw[4], rw[5], rw[6], rw[7]);
}

// ---------------------------------------------------------------------------
// Pack B-matrix (weights, K x Ncols f32) into WMMA 16-bit B 32x16 layout:
//   lane l holds col (l%16); element j -> K = (l/16)*8 + (j<8 ? j : j+8)
// Packed offset for thread t: t = (nb*numKb + kb)*32 + lane
// ---------------------------------------------------------------------------
__global__ void pack_w_kernel(const float* __restrict__ W,
                              unsigned short* __restrict__ outp,
                              int K, int Ncols) {
  int numKb = K >> 5;
  int numNb = Ncols >> 4;
  unsigned t = blockIdx.x * blockDim.x + threadIdx.x;
  unsigned total = (unsigned)numNb * numKb * 32u;
  if (t >= total) return;
  unsigned lane = t & 31u;
  unsigned kb = (t >> 5) % (unsigned)numKb;
  unsigned nb = t / (32u * numKb);
  unsigned half = lane >> 4;
  unsigned n = nb * 16u + (lane & 15u);

  unsigned short o[16];
#pragma unroll
  for (int j = 0; j < 16; ++j) {
    unsigned k = kb * 32u + half * 8u + (unsigned)((j < 8) ? j : j + 8);
    o[j] = f2bf(W[(size_t)k * Ncols + n]);
  }
  unsigned rw[8];
#pragma unroll
  for (int j = 0; j < 8; ++j)
    rw[j] = (unsigned)o[2 * j] | ((unsigned)o[2 * j + 1] << 16);
  uint4* dst4 = reinterpret_cast<uint4*>(outp + (size_t)t * 16);
  dst4[0] = make_uint4(rw[0], rw[1], rw[2], rw[3]);
  dst4[1] = make_uint4(rw[4], rw[5], rw[6], rw[7]);
}

// ---------------------------------------------------------------------------
// WMMA GEMM: out[N x COLS] = A[N x (NUMKB*32)] @ W + bias, optional ReLU +
// deterministic dropout (p=0.5, scale 2x). One block = 16 rows x COLS cols,
// COLS/16 waves; wave w owns columns [16w, 16w+16). 8 WMMAs per wave (K=256).
// ---------------------------------------------------------------------------
template <int NUMKB, int COLS, bool ACT>
__global__ void gemm_wmma_kernel(const unsigned short* __restrict__ A,
                                 const unsigned short* __restrict__ Wp,
                                 const float* __restrict__ bias,
                                 float* __restrict__ outp,
                                 int N, unsigned seed) {
  int lane = threadIdx.x & 31;
  int wave = threadIdx.x >> 5;           // 0 .. COLS/16-1
  int mb = blockIdx.x;

  const v16u* Ap = reinterpret_cast<const v16u*>(A) + (size_t)mb * NUMKB * 32 + lane;
  const v16u* Bp = reinterpret_cast<const v16u*>(Wp) + (size_t)wave * NUMKB * 32 + lane;

  v8f c = {};
#pragma unroll
  for (int kb = 0; kb < NUMKB; ++kb) {
    v16u au = Ap[kb * 32];
    v16u bu = Bp[kb * 32];
    v16bf a = __builtin_bit_cast(v16bf, au);
    v16bf b = __builtin_bit_cast(v16bf, bu);
    c = __builtin_amdgcn_wmma_f32_16x16x32_bf16(
        /*neg_a=*/false, a, /*neg_b=*/false, b,
        /*c_mod=*/(short)0, c, /*reuse_a=*/false, /*reuse_b=*/false);
  }

  int half = lane >> 4;
  int col = wave * 16 + (lane & 15);
  float bv = bias[col];
#pragma unroll
  for (int r = 0; r < 8; ++r) {
    int row = mb * 16 + half * 8 + r;      // C/D layout: VGPR r -> M = half*8 + r
    if (row < N) {
      float v = c[r] + bv;
      if (ACT) {
        v = fmaxf(v, 0.0f);
        unsigned idx = (unsigned)row * (unsigned)COLS + (unsigned)col;
        unsigned hsh = hash32(idx ^ seed);
        v = (hsh & 1u) ? v * 2.0f : 0.0f;   // p = 0.5, scale 1/(1-p)
      }
      outp[(size_t)row * COLS + col] = v;
    }
  }
}

// ---------------------------------------------------------------------------
// Host launcher
// ---------------------------------------------------------------------------
static inline size_t align_up(size_t x, size_t a) { return (x + a - 1) & ~(a - 1); }

extern "C" void kernel_launch(void* const* d_in, const int* in_sizes, int n_in,
                              void* d_out, int out_size, void* d_ws, size_t ws_size,
                              hipStream_t stream) {
  (void)n_in; (void)out_size; (void)ws_size;

  const float* x      = (const float*)d_in[0];
  const int*   edge   = (const int*)d_in[1];
  const float* w[3]   = {(const float*)d_in[2], (const float*)d_in[4], (const float*)d_in[6]};
  const float* bvec[3]= {(const float*)d_in[3], (const float*)d_in[5], (const float*)d_in[7]};
  const float* lin_w  = (const float*)d_in[8];
  const float* lin_b  = (const float*)d_in[9];

  const int N = in_sizes[0] / 128;
  const int E = in_sizes[1] / 2;
  const int* src = edge;
  const int* dst = edge + E;
  const int Mb = (N + 15) >> 4;

  // ---- workspace carve ----
  char* p = (char*)d_ws;
  size_t hbytes = (size_t)N * 128 * sizeof(float);
  float* hA   = (float*)p;  p += align_up(hbytes, 256);
  float* hB   = (float*)p;  p += align_up(hbytes, 256);
  float* agg  = (float*)p;  p += align_up(hbytes, 256);
  float* degf = (float*)p;  p += align_up((size_t)N * sizeof(float), 256);
  unsigned short* Abf = (unsigned short*)p;
  p += align_up((size_t)Mb * 16 * 256 * sizeof(unsigned short), 256);
  unsigned short* wp[3];
  for (int l = 0; l < 3; ++l) { wp[l] = (unsigned short*)p; p += align_up((size_t)256 * 128 * 2, 256); }
  unsigned short* wpl = (unsigned short*)p; p += align_up((size_t)128 * 64 * 2, 256);

  // ---- degrees (shared by all layers) ----
  hipMemsetAsync(degf, 0, (size_t)N * sizeof(float), stream);
  deg_kernel<<<(E + 255) / 256, 256, 0, stream>>>(dst, degf, E);

  // ---- pack weights to bf16 fragment layout (tiny) ----
  for (int l = 0; l < 3; ++l) {
    int total = (128 / 16) * (256 / 32) * 32;   // 2048
    pack_w_kernel<<<(total + 255) / 256, 256, 0, stream>>>(w[l], wp[l], 256, 128);
  }
  {
    int total = (64 / 16) * (128 / 32) * 32;    // 512
    pack_w_kernel<<<(total + 255) / 256, 256, 0, stream>>>(lin_w, wpl, 128, 64);
  }

  // ---- 3 SAGE layers ----
  const float* hcur = x;
  float* houts[3] = {hA, hB, hA};
  unsigned scatterThreads = (unsigned)E * 32u;
  int packTotal = Mb * 8 * 32;
  for (int l = 0; l < 3; ++l) {
    hipMemsetAsync(agg, 0, hbytes, stream);
    scatter_kernel<<<(scatterThreads + 255) / 256, 256, 0, stream>>>(hcur, src, dst, agg, E);
    pack_a_kernel<<<(packTotal + 255) / 256, 256, 0, stream>>>(agg, degf, hcur, Abf, N, 8, 0);
    gemm_wmma_kernel<8, 128, true><<<Mb, 256, 0, stream>>>(
        Abf, wp[l], bvec[l], houts[l], N, 0x9E3779B9u * (unsigned)(l + 1));
    hcur = houts[l];
  }

  // ---- linear head: out = h @ lin_w + lin_b ----
  int packTotalHead = Mb * 4 * 32;
  pack_a_kernel<<<(packTotalHead + 255) / 256, 256, 0, stream>>>(nullptr, nullptr, hcur, Abf, N, 4, 1);
  gemm_wmma_kernel<4, 64, false><<<Mb, 128, 0, stream>>>(
      Abf, wpl, lin_b, (float*)d_out, N, 0u);
}
